// SurrogateDistillClassifierVar_27625229648169
// MI455X (gfx1250) — matrix operand order repaired and checked by reading, not compiled, not run on hardware
//
#include <hip/hip_runtime.h>
#include <math.h>
#include <stdint.h>

// Problem constants (from reference)
#define BB 16384
#define DD 1024
#define OO 2048
#define MU 0.9f
#define LAM 0.1f

// ---------------------------------------------------------------- types
typedef __bf16 bf16_t;
typedef bf16_t v16bf __attribute__((ext_vector_type(16)));
typedef float  v8f   __attribute__((ext_vector_type(8)));
typedef unsigned int v8u __attribute__((ext_vector_type(8)));

__device__ __forceinline__ unsigned short f2bf(float f) {
    unsigned int u = __float_as_uint(f);
    unsigned int r = 0x7FFFu + ((u >> 16) & 1u);
    return (unsigned short)((u + r) >> 16);
}

__device__ __forceinline__ v8f vzero8() { v8f v; for (int i = 0; i < 8; ++i) v[i] = 0.0f; return v; }

// ---------------------------------------------------------------- CDNA5 async global->LDS copy
// GLOBAL_LOAD_ASYNC_TO_LDS_B128 (GV mode): VDST = LDS byte offset (generic LDS addr
// low 32 bits, per flat-aperture rule), VADDR = 64-bit global address. Tracked by
// ASYNCcnt; async loads retire in order, so s_wait_asynccnt N after issuing N newer
// copies guarantees the previous panel is resident.
__device__ __forceinline__ void async_copy_b128(void* lds, const void* gmem) {
    unsigned ldsoff = (unsigned)(uintptr_t)lds;
    asm volatile("global_load_async_to_lds_b128 %0, %1, off"
                 :: "v"(ldsoff), "v"(gmem) : "memory");
}
template <int N>
__device__ __forceinline__ void wait_asynccnt() {
    asm volatile("s_wait_asynccnt %0" :: "i"(N) : "memory");
}

// ---------------------------------------------------------------- WMMA fragment loaders
// CDNA5 16-bit A 16x32 layout: lanes 0-15 row M=lane, VGPR0-3 K=0..7, VGPR4-7 K=16..23;
// lanes 16-31 same rows, K=8..15 and K=24..31. -> two 16B contiguous runs per lane.
__device__ __forceinline__ v16bf frag_A(const unsigned short* lds, int ld, int row0, int lane) {
    int r = lane & 15, hi = lane >> 4;
    const unsigned short* p0 = lds + (size_t)(row0 + r) * ld + hi * 8;
    uint4 a = *(const uint4*)p0;
    uint4 b = *(const uint4*)(p0 + 16);
    v8u f; f[0]=a.x; f[1]=a.y; f[2]=a.z; f[3]=a.w; f[4]=b.x; f[5]=b.y; f[6]=b.z; f[7]=b.w;
    return __builtin_bit_cast(v16bf, f);
}
// 16-bit B 32x16 (stored as [N rows][K] row-major = W[O,D]): lane col N=lane&15,
// lanes 0-15 hold K=0..7,16..23, lanes 16-31 hold K=8..15,24..31 -> two 16B runs per lane.
__device__ __forceinline__ v16bf frag_B(const unsigned short* lds, int ld, int n0, int lane) {
    int c = lane & 15, hi = lane >> 4;
    const unsigned short* p = lds + (size_t)(n0 + c) * ld + hi * 8;
    uint4 a = *(const uint4*)p;
    uint4 b = *(const uint4*)(p + 16);
    v8u f; f[0]=a.x; f[1]=a.y; f[2]=a.z; f[3]=a.w; f[4]=b.x; f[5]=b.y; f[6]=b.z; f[7]=b.w;
    return __builtin_bit_cast(v16bf, f);
}

__device__ __forceinline__ v8f wmma_bf16(v16bf a, v16bf b, v8f c) {
    return __builtin_amdgcn_wmma_f32_16x16x32_bf16(false, a, false, b, (short)0, c, false, false);
}

// ---------------------------------------------------------------- small helpers
__device__ __forceinline__ float block_reduce_sum(float v, float* red) {
    int tid = threadIdx.x;
    red[tid] = v; __syncthreads();
    for (int s = 128; s > 0; s >>= 1) { if (tid < s) red[tid] += red[tid + s]; __syncthreads(); }
    float r = red[0]; __syncthreads();
    return r;
}
__device__ __forceinline__ float block_reduce_max(float v, float* red) {
    int tid = threadIdx.x;
    red[tid] = v; __syncthreads();
    for (int s = 128; s > 0; s >>= 1) { if (tid < s) red[tid] = fmaxf(red[tid], red[tid + s]); __syncthreads(); }
    float r = red[0]; __syncthreads();
    return r;
}

// ---------------------------------------------------------------- K1: segment sums
__global__ __launch_bounds__(256) void segsum_kernel(const float* __restrict__ x,
                                                     const int* __restrict__ labels,
                                                     float* __restrict__ sums,
                                                     float* __restrict__ counts) {
    int b = blockIdx.x;
    int lbl = labels[b];
    if (lbl < 0 || lbl >= OO) return;
    const float* xr = x + (size_t)b * DD;
    float* sr = sums + (size_t)lbl * DD;
    for (int d = threadIdx.x; d < DD; d += 256) atomicAdd(&sr[d], xr[d]);
    if (threadIdx.x == 0) atomicAdd(&counts[lbl], 1.0f);
}

// ---------------------------------------------------------------- K2: rdotc[o] = <r_sub[o], classifier[o]>
__global__ __launch_bounds__(256) void rdotc_kernel(const float* __restrict__ sums,
                                                    const float* __restrict__ counts,
                                                    const float* __restrict__ hist_in,
                                                    const int* __restrict__ hist_cnt,
                                                    const float* __restrict__ none_t,
                                                    const float* __restrict__ clsf,
                                                    float* __restrict__ rdotc) {
    __shared__ float red[256];
    int o = blockIdx.x;
    float part = 0.0f;
    if (o == 0) {
        for (int d = threadIdx.x; d < DD; d += 256) part += none_t[d] * clsf[d];
    } else {
        int h = o - 1;
        float cf = counts[o];
        bool has = cf > 0.0f;
        float cnt = (float)hist_cnt[h] + (has ? 1.0f : 0.0f);
        float denom = 1.0f - powf(MU, fmaxf(cnt, 1.0f));
        float scale = (1.0f - MU) / denom;
        float inv = 1.0f / fmaxf(cf, 1.0f);
        const float* hr = hist_in + (size_t)h * DD;
        const float* sr = sums + (size_t)o * DD;
        const float* cr = clsf + (size_t)o * DD;
        for (int d = threadIdx.x; d < DD; d += 256) {
            float hv = hr[d];
            float nh = has ? (MU * hv + sr[d] * inv) : hv;
            part += nh * scale * cr[d];
        }
    }
    float tot = block_reduce_sum(part, red);
    if (threadIdx.x == 0) rdotc[o] = tot;
}

// ---------------------------------------------------------------- K3: fp32 -> bf16
__global__ __launch_bounds__(256) void cvt_kernel(const float* __restrict__ in,
                                                  unsigned short* __restrict__ out, int n) {
    int i = blockIdx.x * 256 + threadIdx.x;
    if (i < n) out[i] = f2bf(in[i]);
}

// ---------------------------------------------------------------- K4: g_word = sigmoid(x @ word_g_w.T + b)
__global__ __launch_bounds__(256) void gword_kernel(const float* __restrict__ x,
                                                    const float* __restrict__ gw,
                                                    const float* __restrict__ gb,
                                                    float* __restrict__ g_word) {
    int wave = threadIdx.x >> 5, lane = threadIdx.x & 31;
    int b = blockIdx.x * 8 + wave;
    const float* xr = x + (size_t)b * DD;
    float s = 0.0f;
    for (int d = lane; d < DD; d += 32) s += xr[d] * gw[d];
    for (int off = 16; off > 0; off >>= 1) s += __shfl_down(s, off, 32);
    if (lane == 0) g_word[b] = 1.0f / (1.0f + expf(-(s + gb[0])));
}

// ---------------------------------------------------------------- K5: h = gelu(x @ w1.T + b1)  (bf16 out)
// Block tile 128x128, 8 waves (4x2), wave tile 32x64, K step 32.
// Double-buffered LDS panels filled by async global->LDS copies (4 per thread per panel).
__global__ __launch_bounds__(256) void gemm_gelu_kernel(const unsigned short* __restrict__ A,  // [M,K] bf16
                                                        const unsigned short* __restrict__ W,  // [N,K] bf16
                                                        const float* __restrict__ bias,        // [N]
                                                        unsigned short* __restrict__ C,        // [M,N] bf16
                                                        int M, int N, int K) {
    __shared__ unsigned short sA[2][128 * 40];
    __shared__ unsigned short sW[2][128 * 40];
    const int tid = threadIdx.x, lane = tid & 31, wave = tid >> 5;
    const int wm = wave >> 1, wn = wave & 1;
    const int m0 = blockIdx.y * 128, n0 = blockIdx.x * 128;

    // staging coords: 512 x uint4 per matrix panel -> chunks tid and tid+256
    const int r0 = tid >> 2,        c0 = (tid & 3) * 8;
    const int r1 = (tid + 256) >> 2, c1 = ((tid + 256) & 3) * 8;
    const unsigned short* gA0 = A + (size_t)(m0 + r0) * K + c0;
    const unsigned short* gA1 = A + (size_t)(m0 + r1) * K + c1;
    const unsigned short* gW0 = W + (size_t)(n0 + r0) * K + c0;
    const unsigned short* gW1 = W + (size_t)(n0 + r1) * K + c1;

    auto stage = [&](int buf, int k0) {
        async_copy_b128(&sA[buf][r0 * 40 + c0], gA0 + k0);
        async_copy_b128(&sA[buf][r1 * 40 + c1], gA1 + k0);
        async_copy_b128(&sW[buf][r0 * 40 + c0], gW0 + k0);
        async_copy_b128(&sW[buf][r1 * 40 + c1], gW1 + k0);
    };

    v8f acc[2][4];
    for (int i = 0; i < 2; ++i) for (int j = 0; j < 4; ++j) acc[i][j] = vzero8();

    stage(0, 0);
    for (int k0 = 0; k0 < K; k0 += 32) {
        const int buf = (k0 >> 5) & 1;
        const bool more = (k0 + 32) < K;
        if (more) { stage(buf ^ 1, k0 + 32); wait_asynccnt<4>(); }
        else      { wait_asynccnt<0>(); }
        __syncthreads();  // current panel resident for all waves
        v16bf af[2], bfd[4];
        for (int i = 0; i < 2; ++i) af[i] = frag_A(sA[buf], 40, wm * 32 + i * 16, lane);
        for (int j = 0; j < 4; ++j) bfd[j] = frag_B(sW[buf], 40, wn * 64 + j * 16, lane);
        for (int i = 0; i < 2; ++i)
            for (int j = 0; j < 4; ++j)
                acc[i][j] = wmma_bf16(af[i], bfd[j], acc[i][j]);
        __syncthreads();  // all reads of this buffer done before it is refilled
    }

    const int nlo = lane & 15, mhi = (lane >> 4) * 8;
    for (int i = 0; i < 2; ++i)
        for (int j = 0; j < 4; ++j) {
            int n = n0 + wn * 64 + j * 16 + nlo;
            int mb = m0 + wm * 32 + i * 16 + mhi;
            float bv = bias[n];
            for (int t = 0; t < 8; ++t) {
                float v = acc[i][j][t] + bv;
                v = 0.5f * v * (1.0f + erff(v * 0.70710678118f));  // exact gelu
                C[(size_t)(mb + t) * N + n] = f2bf(v);
            }
        }
}

// ---------------------------------------------------------------- K6: fused triple GEMM + epilogue
// logits = (1-w)*(x@cls.T) + w*rdotc ;  w = g_word*sigmoid(x_ctx@q.T+qb) ; mse += (ctx-att)^2
// Block tile 64(M=batch) x 128(N=O), 8 waves (2x4), wave tile 32x32, K step 32.
// Double-buffered LDS, 9 async b128 copies per thread per panel.
__global__ __launch_bounds__(256) void fused_main_kernel(const unsigned short* __restrict__ Xb,   // [B,D]
                                                         const unsigned short* __restrict__ XCb,  // [B,D]
                                                         const unsigned short* __restrict__ Hb,   // [B,D]
                                                         const unsigned short* __restrict__ Cb,   // [O,D] classifier
                                                         const unsigned short* __restrict__ Qb,   // [O,D] ctx_q_w
                                                         const unsigned short* __restrict__ W2b,  // [O,D] tcq_w2
                                                         const float* __restrict__ ctx_b,
                                                         const float* __restrict__ att_b,
                                                         const float* __restrict__ g_word,
                                                         const float* __restrict__ rdotc,
                                                         const int* __restrict__ labels,
                                                         float* __restrict__ logits,
                                                         float* __restrict__ mse_acc) {
    __shared__ unsigned short sX[2][64 * 40], sC[2][64 * 40], sH[2][64 * 40];
    __shared__ unsigned short sWc[2][128 * 40], sWq[2][128 * 40], sW2[2][128 * 40];
    const int tid = threadIdx.x, lane = tid & 31, wave = tid >> 5;
    const int wm = wave >> 2, wn = wave & 3;
    const int m0 = blockIdx.y * 64, n0 = blockIdx.x * 128;

    // A panels: 256 x uint4 (1 chunk/thread); B panels: 512 x uint4 (2 chunks/thread)
    const int ra = tid >> 2,         ca = (tid & 3) * 8;
    const int rb = (tid + 256) >> 2, cb = ((tid + 256) & 3) * 8;
    const size_t gaoff = (size_t)(m0 + ra) * DD + ca;
    const unsigned short* gX  = Xb  + gaoff;
    const unsigned short* gC  = XCb + gaoff;
    const unsigned short* gH  = Hb  + gaoff;
    const size_t gb0 = (size_t)(n0 + ra) * DD + ca;
    const size_t gb1 = (size_t)(n0 + rb) * DD + cb;
    const unsigned short* gWc0 = Cb  + gb0; const unsigned short* gWc1 = Cb  + gb1;
    const unsigned short* gWq0 = Qb  + gb0; const unsigned short* gWq1 = Qb  + gb1;
    const unsigned short* gW20 = W2b + gb0; const unsigned short* gW21 = W2b + gb1;

    auto stage = [&](int buf, int k0) {
        async_copy_b128(&sX[buf][ra * 40 + ca], gX + k0);
        async_copy_b128(&sC[buf][ra * 40 + ca], gC + k0);
        async_copy_b128(&sH[buf][ra * 40 + ca], gH + k0);
        async_copy_b128(&sWc[buf][ra * 40 + ca], gWc0 + k0);
        async_copy_b128(&sWc[buf][rb * 40 + cb], gWc1 + k0);
        async_copy_b128(&sWq[buf][ra * 40 + ca], gWq0 + k0);
        async_copy_b128(&sWq[buf][rb * 40 + cb], gWq1 + k0);
        async_copy_b128(&sW2[buf][ra * 40 + ca], gW20 + k0);
        async_copy_b128(&sW2[buf][rb * 40 + cb], gW21 + k0);
    };

    v8f aC[2][2], aQ[2][2], aA[2][2];
    for (int i = 0; i < 2; ++i) for (int j = 0; j < 2; ++j) { aC[i][j] = vzero8(); aQ[i][j] = vzero8(); aA[i][j] = vzero8(); }

    stage(0, 0);
    for (int k0 = 0; k0 < DD; k0 += 32) {
        const int buf = (k0 >> 5) & 1;
        const bool more = (k0 + 32) < DD;
        if (more) { stage(buf ^ 1, k0 + 32); wait_asynccnt<9>(); }
        else      { wait_asynccnt<0>(); }
        __syncthreads();
        v16bf ax[2], ac[2], ah[2], bc[2], bq[2], b2[2];
        for (int i = 0; i < 2; ++i) {
            ax[i] = frag_A(sX[buf], 40, wm * 32 + i * 16, lane);
            ac[i] = frag_A(sC[buf], 40, wm * 32 + i * 16, lane);
            ah[i] = frag_A(sH[buf], 40, wm * 32 + i * 16, lane);
        }
        for (int j = 0; j < 2; ++j) {
            bc[j] = frag_B(sWc[buf], 40, wn * 32 + j * 16, lane);
            bq[j] = frag_B(sWq[buf], 40, wn * 32 + j * 16, lane);
            b2[j] = frag_B(sW2[buf], 40, wn * 32 + j * 16, lane);
        }
        for (int i = 0; i < 2; ++i)
            for (int j = 0; j < 2; ++j) {
                aC[i][j] = wmma_bf16(ax[i], bc[j], aC[i][j]);
                aQ[i][j] = wmma_bf16(ac[i], bq[j], aQ[i][j]);
                aA[i][j] = wmma_bf16(ah[i], b2[j], aA[i][j]);
            }
        __syncthreads();
    }

    // epilogue
    const int nlo = lane & 15, mhi = (lane >> 4) * 8;
    float gv[2][8], mk[2][8];
    for (int i = 0; i < 2; ++i) {
        int mb = m0 + wm * 32 + i * 16 + mhi;
        for (int t = 0; t < 8; ++t) {
            gv[i][t] = g_word[mb + t];
            mk[i][t] = (labels[mb + t] >= 0) ? 1.0f : 0.0f;
        }
    }
    float msep = 0.0f;
    for (int i = 0; i < 2; ++i)
        for (int j = 0; j < 2; ++j) {
            int n = n0 + wn * 32 + j * 16 + nlo;
            int mb = m0 + wm * 32 + i * 16 + mhi;
            float cbv = ctx_b[n], abv = att_b[n], rd = rdotc[n];
            for (int t = 0; t < 8; ++t) {
                float xc  = aC[i][j][t];
                float ctx = aQ[i][j][t] + cbv;
                float att = aA[i][j][t] + abv;
                float w   = gv[i][t] * (1.0f / (1.0f + expf(-ctx)));
                logits[(size_t)(mb + t) * OO + n] = (1.0f - w) * xc + w * rd;
                float d = ctx - att;
                msep += mk[i][t] * d * d;
            }
        }
    for (int off = 16; off > 0; off >>= 1) msep += __shfl_down(msep, off, 32);
    if (lane == 0) atomicAdd(mse_acc, msep);
}

// ---------------------------------------------------------------- K7: cross-entropy (row log-softmax)
__global__ __launch_bounds__(256) void ce_kernel(const float* __restrict__ logits,
                                                 const int* __restrict__ labels,
                                                 float* __restrict__ ce_acc,
                                                 float* __restrict__ n_acc) {
    __shared__ float red[256];
    int b = blockIdx.x;
    const float* row = logits + (size_t)b * OO;
    float mx = -3.4e38f;
    for (int o = threadIdx.x; o < OO; o += 256) mx = fmaxf(mx, row[o]);
    mx = block_reduce_max(mx, red);
    float s = 0.0f;
    for (int o = threadIdx.x; o < OO; o += 256) s += expf(row[o] - mx);
    s = block_reduce_sum(s, red);
    if (threadIdx.x == 0) {
        int lbl = labels[b];
        if (lbl >= 0) {
            float nll = logf(s) + mx - row[lbl];
            atomicAdd(ce_acc, nll);
            atomicAdd(n_acc, 1.0f);
        }
    }
}

// ---------------------------------------------------------------- K8: finalize scalar loss
__global__ void finalize_kernel(const float* __restrict__ scal, float* __restrict__ out) {
    float n = fmaxf(scal[2], 1.0f);
    out[0] = scal[0] / n + LAM * (scal[1] * (1.0f / (float)OO)) / n;
}

// ---------------------------------------------------------------- host
extern "C" void kernel_launch(void* const* d_in, const int* in_sizes, int n_in,
                              void* d_out, int out_size, void* d_ws, size_t ws_size,
                              hipStream_t stream) {
    const float* x      = (const float*)d_in[0];
    const float* x_ctx  = (const float*)d_in[1];
    const float* clsf   = (const float*)d_in[2];
    const float* none_t = (const float*)d_in[3];
    const float* qw     = (const float*)d_in[4];
    const float* qb     = (const float*)d_in[5];
    const float* gww    = (const float*)d_in[6];
    const float* gwb    = (const float*)d_in[7];
    const float* w1     = (const float*)d_in[8];
    const float* b1     = (const float*)d_in[9];
    const float* w2     = (const float*)d_in[10];
    const float* b2     = (const float*)d_in[11];
    const float* hist   = (const float*)d_in[12];
    const int*   hcnt   = (const int*)d_in[13];
    const int*   labels = (const int*)d_in[14];

    char* ws = (char*)d_ws;
    size_t off = 0;
    auto alloc = [&](size_t bytes) -> void* {
        void* p = ws + off;
        off = (off + bytes + 255) & ~(size_t)255;
        return p;
    };
    // zero-initialized region first (one memset)
    float* sums   = (float*)alloc((size_t)OO * DD * 4);
    float* counts = (float*)alloc((size_t)OO * 4);
    float* scal   = (float*)alloc(64);             // [0]=ce_sum [1]=mse_sum [2]=n
    size_t zbytes = off;
    float* rdotc  = (float*)alloc((size_t)OO * 4);
    float* gword  = (float*)alloc((size_t)BB * 4);
    unsigned short* xb  = (unsigned short*)alloc((size_t)BB * DD * 2);
    unsigned short* xcb = (unsigned short*)alloc((size_t)BB * DD * 2);
    unsigned short* hb  = (unsigned short*)alloc((size_t)BB * DD * 2);
    unsigned short* cb  = (unsigned short*)alloc((size_t)OO * DD * 2);
    unsigned short* qwb = (unsigned short*)alloc((size_t)OO * DD * 2);
    unsigned short* w1b = (unsigned short*)alloc((size_t)DD * DD * 2);
    unsigned short* w2b = (unsigned short*)alloc((size_t)OO * DD * 2);

    float* logits = (float*)d_out + 1;  // d_out = [loss, logits(B*O)]

    hipMemsetAsync(ws, 0, zbytes, stream);

    segsum_kernel<<<BB, 256, 0, stream>>>(x, labels, sums, counts);
    rdotc_kernel<<<OO, 256, 0, stream>>>(sums, counts, hist, hcnt, none_t, clsf, rdotc);

    const int nBD = BB * DD, nOD = OO * DD, nDD = DD * DD;
    cvt_kernel<<<(nBD + 255) / 256, 256, 0, stream>>>(x,     xb,  nBD);
    cvt_kernel<<<(nBD + 255) / 256, 256, 0, stream>>>(x_ctx, xcb, nBD);
    cvt_kernel<<<(nOD + 255) / 256, 256, 0, stream>>>(clsf,  cb,  nOD);
    cvt_kernel<<<(nOD + 255) / 256, 256, 0, stream>>>(qw,    qwb, nOD);
    cvt_kernel<<<(nDD + 255) / 256, 256, 0, stream>>>(w1,    w1b, nDD);
    cvt_kernel<<<(nOD + 255) / 256, 256, 0, stream>>>(w2,    w2b, nOD);

    gword_kernel<<<BB / 8, 256, 0, stream>>>(x, gww, gwb, gword);

    // h = gelu(x @ w1.T + b1): M=BB, N=DD, K=DD
    dim3 gh(DD / 128, BB / 128);
    gemm_gelu_kernel<<<gh, 256, 0, stream>>>(xb, w1b, b1, hb, BB, DD, DD);

    // fused logits + mse
    dim3 gf(OO / 128, BB / 64);
    fused_main_kernel<<<gf, 256, 0, stream>>>(xb, xcb, hb, cb, qwb, w2b,
                                              qb, b2, gword, rdotc, labels,
                                              logits, &scal[1]);

    ce_kernel<<<BB, 256, 0, stream>>>(logits, labels, &scal[0], &scal[2]);
    finalize_kernel<<<1, 1, 0, stream>>>(scal, (float*)d_out);
}
